// InvariantFeGdBFieldModel_83743272337604
// MI455X (gfx1250) — matrix operand order repaired and checked by reading, not compiled, not run on hardware
//
#include <hip/hip_runtime.h>
#include <hip/hip_bf16.h>

#define HDIM   128
#define NNODE  100000
#define NEDGE  1600000
#define NLAYER 3

typedef __attribute__((ext_vector_type(16))) __bf16 vbf16x16;
typedef __attribute__((ext_vector_type(8)))  __bf16 vbf16x8;
typedef __attribute__((ext_vector_type(8)))  float  vf32x8;

#if defined(__gfx1250__) && __has_builtin(__builtin_amdgcn_tensor_load_to_lds) && __has_builtin(__builtin_amdgcn_s_wait_tensorcnt)
#define USE_TDM 1
#endif

#ifdef USE_TDM
typedef unsigned int u32x4 __attribute__((ext_vector_type(4)));
typedef int          i32x8 __attribute__((ext_vector_type(8)));
typedef int          i32x4 __attribute__((ext_vector_type(4)));

// 1-D contiguous global->LDS DMA via Tensor Data Mover (D# per CDNA5 ISA 8.3/8.4).
// bytes must be a multiple of 8. LDS address = low 32 bits of flat shared addr.
// Toolchain carries the 6-arg builtin: (g0, g1, g2, g3, g4, cpol).
__device__ __forceinline__ void tdm_copy_to_lds(const void* gsrc, void* lds, unsigned bytes)
{
    unsigned long long ga = (unsigned long long)(uintptr_t)gsrc;
    unsigned laddr = (unsigned)(uintptr_t)lds;
    unsigned n8 = bytes >> 3;                       // 8-byte elements
    u32x4 g0;
    g0[0] = 1u;                                     // count=1, user descriptor
    g0[1] = laddr;                                  // lds_addr
    g0[2] = (unsigned)(ga & 0xFFFFFFFFu);           // global_addr[31:0]
    g0[3] = (unsigned)((ga >> 32) & 0x1FFFFFFu)     // global_addr[56:32]
          | (2u << 30);                             // type = 2 ("image")
    i32x8 g1;
    g1[0] = (3 << 16);                              // data_size=3 (8B), mask=0
    g1[1] = (int)((n8 & 0xFFFFu) << 16);            // tensor_dim0[15:0]
    g1[2] = (int)((n8 >> 16) & 0xFFFFu) | (1 << 16);// tensor_dim0[31:16] | tensor_dim1=1
    g1[3] = (int)((n8 & 0xFFFFu) << 16);            // tile_dim0 = n8
    g1[4] = 1;                                      // tile_dim1 = 1, tile_dim2 = 0
    g1[5] = (int)n8;                                // tensor_dim0_stride[31:0]
    g1[6] = 0;
    g1[7] = 0;
    i32x4 z4; z4[0] = 0; z4[1] = 0; z4[2] = 0; z4[3] = 0;
    i32x8 z8;
#pragma unroll
    for (int i = 0; i < 8; ++i) z8[i] = 0;
    __builtin_amdgcn_tensor_load_to_lds(g0, g1, z4, z4, z8, 0);
}
#endif

// Block-cooperative load of pre-swizzled bf16 weight fragments into LDS.
// Caller must __syncthreads() afterwards.
__device__ __forceinline__ void load_weights_lds(__bf16* dst, const __bf16* src,
                                                 int nelem, int tid)
{
#ifdef USE_TDM
    if (tid < 32) {                                  // wave 0 issues the DMA
        tdm_copy_to_lds(src, dst, (unsigned)nelem * 2u);
        __builtin_amdgcn_s_wait_tensorcnt(0);
    }
#else
    for (int i = tid; i < (nelem >> 4); i += 128)
        ((vbf16x16*)dst)[i] = ((const vbf16x16*)src)[i];
#endif
}

__device__ __forceinline__ float silu_f(float x) {
    return x * __builtin_amdgcn_rcpf(1.0f + __expf(-x));
}

__device__ __forceinline__ vbf16x16 frag_cat(vbf16x8 lo, vbf16x8 hi) {
    vbf16x16 r;
#pragma unroll
    for (int i = 0; i < 8; ++i) { r[i] = lo[i]; r[i + 8] = hi[i]; }
    return r;
}

__device__ __forceinline__ vbf16x16 frag_from_f32row(const float* rp, int kh) {
    vbf16x16 r;
#pragma unroll
    for (int i = 0; i < 8; ++i) {
        r[i]     = (__bf16)rp[kh * 8 + i];
        r[i + 8] = (__bf16)rp[16 + kh * 8 + i];
    }
    return r;
}

// ---------------------------------------------------------------------------
// Kernel 0: one-time weight pre-swizzle: f32 row-major -> bf16 WMMA B-fragment
// panels. One block per K=32 panel. dstF[(nt*32+lane)*16 + e] layout.
// ---------------------------------------------------------------------------
__global__ void __launch_bounds__(128)
k_wprep(const float* __restrict__ src, __bf16* __restrict__ dst, int krows_total)
{
    int p = blockIdx.x;
    for (int idx = threadIdx.x; idx < 32 * HDIM; idx += 128) {
        int k = idx >> 7;
        int n = idx & 127;
        int ksrc = p * 32 + k;
        float v = (ksrc < krows_total) ? src[(size_t)ksrc * HDIM + n] : 0.0f;
        int nt   = n >> 4;
        int lane = (n & 15) | ((k & 8) << 1);
        int e    = (k & 7) | ((k >> 1) & 8);
        dst[(size_t)p * 4096 + (nt * 32 + lane) * 16 + e] = (__bf16)v;
    }
}

// ---------------------------------------------------------------------------
// Kernel 1: node preprocess. spins normalized, h0 = silu([nt, |s|] @ in_w + b)
// ---------------------------------------------------------------------------
__global__ void k_pre(const float* __restrict__ x,
                      const float* __restrict__ in_w, const float* __restrict__ in_b,
                      float* __restrict__ spins, float* __restrict__ h)
{
    int t = blockIdx.x * blockDim.x + threadIdx.x;   // node*128 + j
    if (t >= NNODE * HDIM) return;
    int node = t >> 7;
    int j    = t & 127;
    const float* xr = x + (size_t)node * 5;
    float nt0 = xr[0], nt1 = xr[1];
    float sx = xr[2], sy = xr[3], sz = xr[4];
    float n0  = sqrtf(sx * sx + sy * sy + sz * sz);
    float inv = __builtin_amdgcn_rcpf(fmaxf(n0, 1e-12f));
    sx *= inv; sy *= inv; sz *= inv;
    float sn = sqrtf(sx * sx + sy * sy + sz * sz);
    if (j < 3) {
        float sv = (j == 0) ? sx : (j == 1) ? sy : sz;
        spins[(size_t)node * 3 + j] = sv;
    }
    float v = nt0 * in_w[0 * HDIM + j] + nt1 * in_w[1 * HDIM + j]
            + sn  * in_w[2 * HDIM + j] + in_b[j];
    h[(size_t)node * HDIM + j] = silu_f(v);
}

// ---------------------------------------------------------------------------
// Kernel 2: per-edge invariant scalar features, bf16, K padded 10 -> 16
// ---------------------------------------------------------------------------
__global__ void k_scal(const int* __restrict__ eidx,      // [2, E]
                       const float* __restrict__ eattr,   // [E, 4]
                       const float* __restrict__ x,
                       const float* __restrict__ spins,
                       __bf16* __restrict__ scal)          // [E, 16]
{
    int e = blockIdx.x * blockDim.x + threadIdx.x;
    if (e >= NEDGE) return;
    int s = eidx[e];
    int d = eidx[NEDGE + e];
    float mjx = spins[(size_t)s * 3 + 0], mjy = spins[(size_t)s * 3 + 1], mjz = spins[(size_t)s * 3 + 2];
    float mix = spins[(size_t)d * 3 + 0], miy = spins[(size_t)d * 3 + 1], miz = spins[(size_t)d * 3 + 2];
    float ux = eattr[(size_t)e * 4 + 0], uy = eattr[(size_t)e * 4 + 1], uz = eattr[(size_t)e * 4 + 2];
    float dist = eattr[(size_t)e * 4 + 3];
    vbf16x16 o;
    o[0] = (__bf16)sqrtf(mix * mix + miy * miy + miz * miz);
    o[1] = (__bf16)sqrtf(mjx * mjx + mjy * mjy + mjz * mjz);
    o[2] = (__bf16)(mix * mjx + miy * mjy + miz * mjz);
    o[3] = (__bf16)(mix * ux + miy * uy + miz * uz);
    o[4] = (__bf16)(mjx * ux + mjy * uy + mjz * uz);
    o[5] = (__bf16)dist;
    o[6] = (__bf16)x[(size_t)d * 5 + 0];
    o[7] = (__bf16)x[(size_t)d * 5 + 1];
    o[8] = (__bf16)x[(size_t)s * 5 + 0];
    o[9] = (__bf16)x[(size_t)s * 5 + 1];
#pragma unroll
    for (int i = 10; i < 16; ++i) o[i] = (__bf16)0.0f;
    *(vbf16x16*)(scal + (size_t)e * 16) = o;   // one 32B store
}

// ---------------------------------------------------------------------------
// Kernel 3: fused edge MLP + scatter-add.  wf = [w1 panel | w2 4 panels] bf16
// fragments (40KB, TDM-copied to LDS in one descriptor).
// ---------------------------------------------------------------------------
__global__ void __launch_bounds__(128)
k_edge(const __bf16* __restrict__ scal,
       const int* __restrict__ eidx,
       const __bf16* __restrict__ wf,
       const float* __restrict__ b1, const float* __restrict__ b2,
       float* __restrict__ aggr)
{
    __shared__ alignas(64) __bf16 sWf[5 * 4096];            // 40 KB: w1f + w2f
    __shared__ float  sB1[HDIM];
    __shared__ float  sB2[HDIM];
    __shared__ alignas(64) __bf16 sStage[4][16 * HDIM];     // 16 KB
    __shared__ int    sDst[4][16];

    int tid = threadIdx.x;
    load_weights_lds(sWf, wf, 5 * 4096, tid);
    if (tid < HDIM) { sB1[tid] = b1[tid]; sB2[tid] = b2[tid]; }

    const __bf16* sW1f = sWf;
    const __bf16* sW2f = sWf + 4096;

    int wave = tid >> 5, lane = tid & 31;
    int m = lane & 15, kh = lane >> 4;
    int ebase = blockIdx.x * 64 + wave * 16;
    if (lane < 16) sDst[wave][lane] = eidx[NEDGE + ebase + lane];
    __syncthreads();

    // --- GEMM1: hid = silu(scalars @ w1 + b1), K=16 of 32 (padding is zero) ---
    vbf16x16 a;
    {
        vbf16x8 lo = *(const vbf16x8*)(scal + (size_t)(ebase + m) * 16 + kh * 8);
#pragma unroll
        for (int i = 0; i < 8; ++i) { a[i] = lo[i]; a[i + 8] = (__bf16)0.0f; }
    }
#pragma unroll
    for (int nt = 0; nt < 8; ++nt) {
        vbf16x16 b = *(const vbf16x16*)(sW1f + (nt * 32 + lane) * 16);
        vf32x8 c = {};
        c = __builtin_amdgcn_wmma_f32_16x16x32_bf16(false, a, false, b, (short)0, c, false, false);
#pragma unroll
        for (int r = 0; r < 8; ++r) {
            int mm  = r + 8 * kh;
            int col = nt * 16 + m;
            sStage[wave][mm * HDIM + col] = (__bf16)silu_f(c[r] + sB1[col]);
        }
    }
    __syncthreads();

    // --- GEMM2: msg = hid @ w2 + b2, scatter-add into aggr[dst] ---
#pragma unroll
    for (int nt = 0; nt < 8; ++nt) {
        vf32x8 c = {};
#pragma unroll
        for (int kt = 0; kt < 4; ++kt) {
            const __bf16* sp = &sStage[wave][m * HDIM + kt * 32 + kh * 8];
            vbf16x16 a2 = frag_cat(*(const vbf16x8*)sp, *(const vbf16x8*)(sp + 16));
            vbf16x16 bv = *(const vbf16x16*)(sW2f + ((kt * 8 + nt) * 32 + lane) * 16);
            c = __builtin_amdgcn_wmma_f32_16x16x32_bf16(false, a2, false, bv, (short)0, c, false, false);
        }
#pragma unroll
        for (int r = 0; r < 8; ++r) {
            int mm  = r + 8 * kh;
            int col = nt * 16 + m;
            unsafeAtomicAdd(&aggr[(size_t)sDst[wave][mm] * HDIM + col], c[r] + sB2[col]);
        }
    }
}

// ---------------------------------------------------------------------------
// Kernel 4: node update. h += silu([h, aggr] @ nw1 + nb1) @ nw2 + nb2
// wf = [nw1 8 panels | nw2 4 panels]; nw1 panels TDM-streamed through LDS.
// ---------------------------------------------------------------------------
__global__ void __launch_bounds__(128)
k_node(float* h,
       const float* __restrict__ aggr,
       const __bf16* __restrict__ wf,
       const float* __restrict__ b1, const float* __restrict__ b2)
{
    __shared__ alignas(64) __bf16 sW2f[4 * 4096];           // 32 KB
    __shared__ alignas(64) __bf16 sWkf[4096];               // 8 KB chunk
    __shared__ float  sB1[HDIM];
    __shared__ float  sB2[HDIM];
    __shared__ alignas(64) __bf16 sStage[4][16 * HDIM];     // 16 KB

    int tid = threadIdx.x;
    load_weights_lds(sW2f, wf + 8 * 4096, 4 * 4096, tid);
    if (tid < HDIM) { sB1[tid] = b1[tid]; sB2[tid] = b2[tid]; }

    int wave = tid >> 5, lane = tid & 31;
    int m = lane & 15, kh = lane >> 4;
    int nbase = blockIdx.x * 64 + wave * 16;
    int row = nbase + m; if (row >= NNODE) row = NNODE - 1;

    vf32x8 zero8 = {};
    vf32x8 acc[8];
#pragma unroll
    for (int nt = 0; nt < 8; ++nt) acc[nt] = zero8;

    for (int kc = 0; kc < 8; ++kc) {            // 8 panels of K=32 (total 256)
        __syncthreads();
        load_weights_lds(sWkf, wf + kc * 4096, 4096, tid);
        __syncthreads();
        const float* src = (kc < 4) ? h : aggr;
        const float* rp = src + (size_t)row * HDIM + (kc & 3) * 32;
        vbf16x16 a = frag_from_f32row(rp, kh);
#pragma unroll
        for (int nt = 0; nt < 8; ++nt) {
            vbf16x16 b = *(const vbf16x16*)(sWkf + (nt * 32 + lane) * 16);
            acc[nt] = __builtin_amdgcn_wmma_f32_16x16x32_bf16(false, a, false, b, (short)0, acc[nt], false, false);
        }
    }
    __syncthreads();
#pragma unroll
    for (int nt = 0; nt < 8; ++nt)
#pragma unroll
        for (int r = 0; r < 8; ++r) {
            int mm = r + 8 * kh, col = nt * 16 + m;
            sStage[wave][mm * HDIM + col] = (__bf16)silu_f(acc[nt][r] + sB1[col]);
        }
    __syncthreads();

#pragma unroll
    for (int nt = 0; nt < 8; ++nt) {
        vf32x8 c = {};
#pragma unroll
        for (int kt = 0; kt < 4; ++kt) {
            const __bf16* sp = &sStage[wave][m * HDIM + kt * 32 + kh * 8];
            vbf16x16 a2 = frag_cat(*(const vbf16x8*)sp, *(const vbf16x8*)(sp + 16));
            vbf16x16 bv = *(const vbf16x16*)(sW2f + ((kt * 8 + nt) * 32 + lane) * 16);
            c = __builtin_amdgcn_wmma_f32_16x16x32_bf16(false, a2, false, bv, (short)0, c, false, false);
        }
#pragma unroll
        for (int r = 0; r < 8; ++r) {
            int mm = r + 8 * kh, col = nt * 16 + m;
            int node = nbase + mm;
            if (node < NNODE) {
                size_t o = (size_t)node * HDIM + col;
                h[o] = h[o] + c[r] + sB2[col];
            }
        }
    }
}

// ---------------------------------------------------------------------------
// Kernel 5: output head. wnode = silu(h @ ow1 + ob1) @ ow2 + ob2 (scalar/node)
// ---------------------------------------------------------------------------
__global__ void __launch_bounds__(128)
k_head(const float* __restrict__ h,
       const __bf16* __restrict__ wf,                       // ow1 4 panels
       const float* __restrict__ ob1,
       const float* __restrict__ ow2, const float* __restrict__ ob2,
       float* __restrict__ wnode)
{
    __shared__ alignas(64) __bf16 sW1f[4 * 4096];           // 32 KB
    __shared__ float  sB1[HDIM];
    __shared__ float  sC[HDIM];
    __shared__ float  sRed[4][16];

    int tid = threadIdx.x;
    load_weights_lds(sW1f, wf, 4 * 4096, tid);
    if (tid < HDIM) { sB1[tid] = ob1[tid]; sC[tid] = ow2[tid]; }

    int wave = tid >> 5, lane = tid & 31;
    int m = lane & 15, kh = lane >> 4;
    if (lane < 16) sRed[wave][lane] = 0.0f;
    int nbase = blockIdx.x * 64 + wave * 16;
    int row = nbase + m; if (row >= NNODE) row = NNODE - 1;
    __syncthreads();

    vf32x8 zero8 = {};
    vf32x8 acc[8];
#pragma unroll
    for (int nt = 0; nt < 8; ++nt) acc[nt] = zero8;
#pragma unroll
    for (int kt = 0; kt < 4; ++kt) {
        const float* rp = h + (size_t)row * HDIM + kt * 32;
        vbf16x16 a = frag_from_f32row(rp, kh);
#pragma unroll
        for (int nt = 0; nt < 8; ++nt) {
            vbf16x16 b = *(const vbf16x16*)(sW1f + ((kt * 8 + nt) * 32 + lane) * 16);
            acc[nt] = __builtin_amdgcn_wmma_f32_16x16x32_bf16(false, a, false, b, (short)0, acc[nt], false, false);
        }
    }
    float psum[8];
#pragma unroll
    for (int r = 0; r < 8; ++r) psum[r] = 0.0f;
#pragma unroll
    for (int nt = 0; nt < 8; ++nt)
#pragma unroll
        for (int r = 0; r < 8; ++r) {
            int col = nt * 16 + m;
            psum[r] += silu_f(acc[nt][r] + sB1[col]) * sC[col];
        }
#pragma unroll
    for (int r = 0; r < 8; ++r)
        atomicAdd(&sRed[wave][r + 8 * kh], psum[r]);   // ds_add_f32
    __syncthreads();
    if (lane < 16) {
        int node = nbase + lane;
        if (node < NNODE) wnode[node] = sRed[wave][lane] + ob2[0];
    }
}

// ---------------------------------------------------------------------------
// Kernel 6: b_field scatter: out[dst] += wnode[src] * spins[src]
// ---------------------------------------------------------------------------
__global__ void k_scatter(const int* __restrict__ eidx,
                          const float* __restrict__ spins,
                          const float* __restrict__ wnode,
                          float* __restrict__ out)
{
    int e = blockIdx.x * blockDim.x + threadIdx.x;
    if (e >= NEDGE) return;
    int s = eidx[e], d = eidx[NEDGE + e];
    float w = wnode[s];
#pragma unroll
    for (int c2 = 0; c2 < 3; ++c2)
        unsafeAtomicAdd(&out[(size_t)d * 3 + c2], w * spins[(size_t)s * 3 + c2]);
}

__global__ void k_zero(float* __restrict__ p, long n)
{
    long i = (long)blockIdx.x * blockDim.x + threadIdx.x;
    long stride = (long)gridDim.x * blockDim.x;
    for (; i < n; i += stride) p[i] = 0.0f;
}

// ---------------------------------------------------------------------------
extern "C" void kernel_launch(void* const* d_in, const int* in_sizes, int n_in,
                              void* d_out, int out_size, void* d_ws, size_t ws_size,
                              hipStream_t stream)
{
    (void)in_sizes; (void)n_in; (void)out_size; (void)ws_size;
    const float* x     = (const float*)d_in[0];
    const int*   eidx  = (const int*)  d_in[1];
    const float* eattr = (const float*)d_in[2];
    const float* in_w  = (const float*)d_in[3];
    const float* in_b  = (const float*)d_in[4];
    const float* ew1   = (const float*)d_in[5];
    const float* eb1   = (const float*)d_in[6];
    const float* ew2   = (const float*)d_in[7];
    const float* eb2   = (const float*)d_in[8];
    const float* nw1   = (const float*)d_in[9];
    const float* nb1   = (const float*)d_in[10];
    const float* nw2   = (const float*)d_in[11];
    const float* nb2   = (const float*)d_in[12];
    const float* ow1   = (const float*)d_in[13];
    const float* ob1   = (const float*)d_in[14];
    const float* ow2   = (const float*)d_in[15];
    const float* ob2   = (const float*)d_in[16];

    char* ws = (char*)d_ws;
    size_t off = 0;
    auto wsa = [&](size_t bytes) -> void* {
        void* p = ws + off;
        off = (off + bytes + 255) & ~(size_t)255;
        return p;
    };
    float*  spins = (float*) wsa((size_t)NNODE * 3 * sizeof(float));
    float*  h     = (float*) wsa((size_t)NNODE * HDIM * sizeof(float));
    float*  aggr  = (float*) wsa((size_t)NNODE * HDIM * sizeof(float));
    float*  wnode = (float*) wsa((size_t)NNODE * sizeof(float));
    __bf16* scal  = (__bf16*)wsa((size_t)NEDGE * 16 * sizeof(__bf16));
    // pre-swizzled weights: per layer edge = 5 panels, node = 12 panels; head = 4
    __bf16* wfrag = (__bf16*)wsa((size_t)(NLAYER * (5 + 12) + 4) * 4096 * sizeof(__bf16));

    // --- one-time weight pre-swizzle into bf16 fragment panels ---
    for (int l = 0; l < NLAYER; ++l) {
        __bf16* we = wfrag + (size_t)l * 5 * 4096;
        __bf16* wn = wfrag + (size_t)(NLAYER * 5 + l * 12) * 4096;
        k_wprep<<<1, 128, 0, stream>>>(ew1 + (size_t)l * 10 * HDIM,   we,            10);
        k_wprep<<<4, 128, 0, stream>>>(ew2 + (size_t)l * HDIM * HDIM, we + 4096,    128);
        k_wprep<<<8, 128, 0, stream>>>(nw1 + (size_t)l * 2 * HDIM * HDIM, wn,       256);
        k_wprep<<<4, 128, 0, stream>>>(nw2 + (size_t)l * HDIM * HDIM, wn + 8 * 4096, 128);
    }
    __bf16* wh = wfrag + (size_t)(NLAYER * 17) * 4096;
    k_wprep<<<4, 128, 0, stream>>>(ow1, wh, 128);

    k_pre <<<(NNODE * HDIM + 255) / 256, 256, 0, stream>>>(x, in_w, in_b, spins, h);
    k_scal<<<(NEDGE + 255) / 256, 256, 0, stream>>>(eidx, eattr, x, spins, scal);

    for (int l = 0; l < NLAYER; ++l) {
        __bf16* we = wfrag + (size_t)l * 5 * 4096;
        __bf16* wn = wfrag + (size_t)(NLAYER * 5 + l * 12) * 4096;
        k_zero<<<2048, 256, 0, stream>>>(aggr, (long)NNODE * HDIM);
        k_edge<<<NEDGE / 64, 128, 0, stream>>>(scal, eidx, we,
            eb1 + (size_t)l * HDIM, eb2 + (size_t)l * HDIM, aggr);
        k_node<<<(NNODE + 63) / 64, 128, 0, stream>>>(h, aggr, wn,
            nb1 + (size_t)l * HDIM, nb2 + (size_t)l * HDIM);
    }

    k_head<<<(NNODE + 63) / 64, 128, 0, stream>>>(h, wh, ob1, ow2, ob2, wnode);
    k_zero<<<512, 256, 0, stream>>>((float*)d_out, (long)NNODE * 3);
    k_scatter<<<(NEDGE + 255) / 256, 256, 0, stream>>>(eidx, spins, wnode, (float*)d_out);
}
